// HeteroGNN3_59966333387121
// MI455X (gfx1250) — matrix operand order repaired and checked by reading, not compiled, not run on hardware
//
#include <hip/hip_runtime.h>
#include <hip/hip_bf16.h>
#include <math.h>

// ---------------- problem constants (match reference) ----------------
#define H   128
#define L   2
#define NF  100000
#define NC  2000
#define B   256
#define DF  1536
#define DC  27
#define EPS 1e-5f

// padded LDS strides (in halfs): stride/2 dwords must be != 0 mod 64; choose == 4 (mod 64)
#define ATILE_LDF 1544   // DF + 8 : 772 dwords = 4 mod 64  -> conflict-free b128 frag reads
#define ATILE_LDH 136    // H  + 8 :  68 dwords = 4 mod 64

// ---------------- CDNA5 WMMA types ----------------
typedef __attribute__((ext_vector_type(16))) __bf16          v16bf;
typedef __attribute__((ext_vector_type(8)))  float           v8f;
typedef __attribute__((ext_vector_type(8)))  unsigned short  ush8;
typedef __attribute__((ext_vector_type(4)))  unsigned short  ush4;

__device__ __forceinline__ unsigned short f2bf(float x) {
    __bf16 b = (__bf16)x;                       // RNE; lets clang use v_cvt_pk_bf16_f32
    return __builtin_bit_cast(unsigned short, b);
}

union FragBF { v16bf v; ush8 h[2]; unsigned short s[16]; };

// A fragment 16x32 (MxK) bf16 from LDS tile (row-major, stride ldh halfs), k order used here (same as B):
//   M = lane%16 ; K runs [k0+(lane/16)*16 .. +15] (contiguous)
__device__ __forceinline__ v16bf load_a_lds(const unsigned short* aT, int ldh,
                                            int k0, int lane) {
    const int m  = lane & 15;
    const int ka = k0 + ((lane >> 4) << 4);
    const unsigned short* p = aT + m * ldh + ka;
    FragBF u;
    u.h[0] = *(const ush8*)(p);
    u.h[1] = *(const ush8*)(p + 8);
    return u.v;
}

// B fragment 32x16 (KxN) with B[k][n] = W[n][k], W preconverted bf16 row-major (ld = K dim):
//   N = lane%16 ; K = k0 + (lane/16)*16 + h, h=0..15 (contiguous 32B in W row)
__device__ __forceinline__ v16bf load_b_bf(const unsigned short* __restrict__ Wbf,
                                           int ld, int n0, int k0, int lane) {
    const int n  = n0 + (lane & 15);
    const int kk = k0 + ((lane >> 4) << 4);
    const unsigned short* p = Wbf + (size_t)n * ld + kk;
    FragBF u;
    u.h[0] = ((const ush8*)p)[0];
    u.h[1] = ((const ush8*)p)[1];
    return u.v;
}

__device__ __forceinline__ float sigmoidf(float x) { return 1.0f / (1.0f + expf(-x)); }

// ---------------- f32 -> bf16 weight preconversion ----------------
__global__ __launch_bounds__(256)
void k_f32_to_bf16(const float* __restrict__ src, unsigned short* __restrict__ dst, int n) {
    const int i = (blockIdx.x * blockDim.x + threadIdx.x) * 4;
    if (i >= n) return;
    const float4 v = *(const float4*)(src + i);
    ush4 o;
    o.x = f2bf(v.x); o.y = f2bf(v.y); o.z = f2bf(v.z); o.w = f2bf(v.w);
    *(ush4*)(dst + i) = o;
}

// ---------------- shared epilogue: bias+relu into LDS, then LayerNorm ----------------
__device__ __forceinline__ void wmma_epilogue_ln(
    float (*tile)[132], const v8f& c, const float* __restrict__ bias,
    const float* __restrict__ gain, const float* __restrict__ beta,
    float* __restrict__ out, int m0) {
    __shared__ float rsum[16][17], rsq[16][17];
    __shared__ float s_mu[16], s_rs[16];
    const int lane  = threadIdx.x & 31;
    const int wv    = threadIdx.x >> 5;
    const int nloc  = (wv << 4) + (lane & 15);
    const int rbase = (lane >> 4) << 3;
    const float bn  = bias[nloc];
#pragma unroll
    for (int v = 0; v < 8; ++v)
        tile[rbase + v][nloc] = fmaxf(c[v] + bn, 0.0f);
    __syncthreads();
    const int r = threadIdx.x >> 4, j = threadIdx.x & 15;
    float s = 0.f, q = 0.f;
#pragma unroll
    for (int cI = j; cI < H; cI += 16) { float v = tile[r][cI]; s += v; q += v * v; }
    rsum[r][j] = s; rsq[r][j] = q;
    __syncthreads();
    if (j == 0) {
        float S = 0.f, Q = 0.f;
#pragma unroll
        for (int i = 0; i < 16; ++i) { S += rsum[r][i]; Q += rsq[r][i]; }
        float mu  = S * (1.0f / H);
        float var = Q * (1.0f / H) - mu * mu;
        s_mu[r] = mu;
        s_rs[r] = rsqrtf(var + EPS);
    }
    __syncthreads();
    const int idx = threadIdx.x * 8;
    const int rr = idx >> 7, c0 = idx & 127;
    const float mu = s_mu[rr], rs = s_rs[rr];
    float* orow = out + (size_t)(m0 + rr) * H;
#pragma unroll
    for (int i = 0; i < 8; ++i) {
        int cc = c0 + i;
        orow[cc] = (tile[rr][cc] - mu) * rs * gain[cc] + beta[cc];
    }
}

// ---------------- kernel 1: fact encoder  hf = LN(relu(x @ W^T + b)) ----------------
// block = 256 thr (8 waves); stages 16xDF x-tile in LDS as bf16 once (each element
// converted exactly once globally), then 48 K-steps of ds-fed WMMA.
__global__ __launch_bounds__(256)
void k_encode_fact(const float* __restrict__ x, const unsigned short* __restrict__ Wbf,
                   const float* __restrict__ bias, const float* __restrict__ gain,
                   const float* __restrict__ beta, float* __restrict__ out) {
    __shared__ unsigned short aT[16 * ATILE_LDF];
    __shared__ float tile[16][132];
    const int lane = threadIdx.x & 31;
    const int wv   = threadIdx.x >> 5;
    const int m0   = blockIdx.x << 4;
    const int n0   = wv << 4;
    // cooperative load + convert: 16*DF contiguous floats starting at x + m0*DF
    const float* xbase = x + (size_t)m0 * DF;
#pragma unroll
    for (int it = 0; it < (16 * DF) / (256 * 4); ++it) {
        const int i = (threadIdx.x + it * 256) * 4;
        const float4 v = *(const float4*)(xbase + i);
        const int r = i / DF, cc = i - r * DF;
        ush4 o;
        o.x = f2bf(v.x); o.y = f2bf(v.y); o.z = f2bf(v.z); o.w = f2bf(v.w);
        *(ush4*)(aT + r * ATILE_LDF + cc) = o;
    }
    __syncthreads();
    v8f c = {};
    for (int k0 = 0; k0 < DF; k0 += 32) {
        v16bf a = load_a_lds(aT, ATILE_LDF, k0, lane);
        v16bf b = load_b_bf(Wbf, DF, n0, k0, lane);
        c = __builtin_amdgcn_wmma_f32_16x16x32_bf16(false, a, false, b, (short)0, c, false, false);
    }
    __syncthreads();
    wmma_epilogue_ln(tile, c, bias, gain, beta, out, m0);
}

// ---------------- kernel 2: comp encoder (DC=27, VALU path, one block per node) ----
__global__ __launch_bounds__(128)
void k_encode_comp(const float* __restrict__ x, const float* __restrict__ W,
                   const float* __restrict__ bias, const float* __restrict__ gain,
                   const float* __restrict__ beta, float* __restrict__ out) {
    __shared__ float r1[128], r2[128];
    const int nid = blockIdx.x, h = threadIdx.x;
    const float* xr = x + (size_t)nid * DC;
    const float* wr = W + (size_t)h * DC;
    float s = bias[h];
#pragma unroll
    for (int d = 0; d < DC; ++d) s += xr[d] * wr[d];
    s = fmaxf(s, 0.0f);
    r1[h] = s; r2[h] = s * s;
    __syncthreads();
    for (int st = 64; st > 0; st >>= 1) {
        if (h < st) { r1[h] += r1[h + st]; r2[h] += r2[h + st]; }
        __syncthreads();
    }
    float mu  = r1[0] * (1.0f / H);
    float var = r2[0] * (1.0f / H) - mu * mu;
    out[(size_t)nid * H + h] = (s - mu) * rsqrtf(var + EPS) * gain[h] + beta[h];
}

// ---------------- kernel 3: learned edge gates ----------------
__global__ __launch_bounds__(256)
void k_edge_gate(const float* __restrict__ ea_fc, const float* __restrict__ ea_cf,
                 const float* __restrict__ wmix, const float* __restrict__ bmix,
                 float* __restrict__ wfc, float* __restrict__ wcf, int e) {
    const int tid = blockIdx.x * blockDim.x + threadIdx.x;
    const float wm = wmix[0], bm = bmix[0];
    if (tid < e)            wfc[tid]     = sigmoidf(ea_fc[tid] * wm + bm);
    else if (tid < 2 * e)   wcf[tid - e] = sigmoidf(ea_cf[tid - e] * wm + bm);
}

// ---------------- kernel 4: edge scatter-add (one wave per edge) ----------------
__global__ __launch_bounds__(256)
void k_scatter(const int* __restrict__ src, const int* __restrict__ dst,
               const float* __restrict__ gate, const float* __restrict__ hsrc,
               float* __restrict__ msg, int e) {
    const int tid  = blockIdx.x * blockDim.x + threadIdx.x;
    const int eid  = tid >> 5;
    const int lane = tid & 31;
    if (eid >= e) return;
    const int s = src[eid], d = dst[eid];
    const float w = gate[eid];
    const float4 v = reinterpret_cast<const float4*>(hsrc + (size_t)s * H)[lane];
    float* mp = msg + (size_t)d * H + lane * 4;
    unsafeAtomicAdd(mp + 0, w * v.x);
    unsafeAtomicAdd(mp + 1, w * v.y);
    unsafeAtomicAdd(mp + 2, w * v.z);
    unsafeAtomicAdd(mp + 3, w * v.w);
}

// ---------------- kernel 5: GraphConv update  LN(relu(msg@Wrel^T + brel + h@Wroot^T)) --
// Stages both 16xH A-tiles (msg + root) in LDS as bf16 once; weights pre-bf16.
__global__ __launch_bounds__(256)
void k_gnn_update(const float* __restrict__ msg, const unsigned short* __restrict__ WrelBf,
                  const float* __restrict__ brel, const float* __restrict__ hroot,
                  const unsigned short* __restrict__ WrootBf, const float* __restrict__ gain,
                  const float* __restrict__ beta, float* __restrict__ out) {
    __shared__ unsigned short aMsg[16 * ATILE_LDH];
    __shared__ unsigned short aRoot[16 * ATILE_LDH];
    __shared__ float tile[16][132];
    const int lane = threadIdx.x & 31;
    const int wv   = threadIdx.x >> 5;
    const int m0   = blockIdx.x << 4;
    const int n0   = wv << 4;
    // cooperative convert: 16*H = 2048 contiguous floats per source, 8 per thread
    {
        const int i = threadIdx.x * 8;              // row r = i>>7, col c0 = i&127
        const int r = i >> 7, c0 = i & 127;
        const float4 v0 = *(const float4*)(msg + (size_t)m0 * H + i);
        const float4 v1 = *(const float4*)(msg + (size_t)m0 * H + i + 4);
        ush8 o;
        o[0] = f2bf(v0.x); o[1] = f2bf(v0.y); o[2] = f2bf(v0.z); o[3] = f2bf(v0.w);
        o[4] = f2bf(v1.x); o[5] = f2bf(v1.y); o[6] = f2bf(v1.z); o[7] = f2bf(v1.w);
        *(ush8*)(aMsg + r * ATILE_LDH + c0) = o;
        const float4 w0 = *(const float4*)(hroot + (size_t)m0 * H + i);
        const float4 w1 = *(const float4*)(hroot + (size_t)m0 * H + i + 4);
        ush8 p;
        p[0] = f2bf(w0.x); p[1] = f2bf(w0.y); p[2] = f2bf(w0.z); p[3] = f2bf(w0.w);
        p[4] = f2bf(w1.x); p[5] = f2bf(w1.y); p[6] = f2bf(w1.z); p[7] = f2bf(w1.w);
        *(ush8*)(aRoot + r * ATILE_LDH + c0) = p;
    }
    __syncthreads();
    v8f c = {};
#pragma unroll
    for (int k0 = 0; k0 < H; k0 += 32) {
        v16bf a0 = load_a_lds(aMsg, ATILE_LDH, k0, lane);
        v16bf b0 = load_b_bf(WrelBf, H, n0, k0, lane);
        c = __builtin_amdgcn_wmma_f32_16x16x32_bf16(false, a0, false, b0, (short)0, c, false, false);
        v16bf a1 = load_a_lds(aRoot, ATILE_LDH, k0, lane);
        v16bf b1 = load_b_bf(WrootBf, H, n0, k0, lane);
        c = __builtin_amdgcn_wmma_f32_16x16x32_bf16(false, a1, false, b1, (short)0, c, false, false);
    }
    __syncthreads();
    wmma_epilogue_ln(tile, c, brel, gain, beta, out, m0);
}

// ---------------- kernel 6a: fact pooling (segment sums) ----------------
__global__ __launch_bounds__(256)
void k_pool_fact(const float* __restrict__ hf, const int* __restrict__ batch,
                 float* __restrict__ fsum, float* __restrict__ cnt) {
    const int tid = blockIdx.x * blockDim.x + threadIdx.x;
    const int n = tid >> 5, lane = tid & 31;
    if (n >= NF) return;
    const int b = batch[n];
    const float4 v = reinterpret_cast<const float4*>(hf + (size_t)n * H)[lane];
    float* dst = fsum + (size_t)b * H + lane * 4;
    unsafeAtomicAdd(dst + 0, v.x);
    unsafeAtomicAdd(dst + 1, v.y);
    unsafeAtomicAdd(dst + 2, v.z);
    unsafeAtomicAdd(dst + 3, v.w);
    if (lane == 0) unsafeAtomicAdd(&cnt[b], 1.0f);
}

// ---------------- kernel 6b: company pooling (masked + fallback sums) ----------------
__global__ __launch_bounds__(256)
void k_pool_comp(const float* __restrict__ hc, const int* __restrict__ batch,
                 const unsigned char* __restrict__ mask,
                 float* __restrict__ smask, float* __restrict__ sall,
                 float* __restrict__ cntm, float* __restrict__ cntc) {
    const int tid = blockIdx.x * blockDim.x + threadIdx.x;
    const int n = tid >> 5, lane = tid & 31;
    if (n >= NC) return;
    const int b = batch[n];
    const float mw = mask[n] ? 1.0f : 0.0f;
    const float4 v = reinterpret_cast<const float4*>(hc + (size_t)n * H)[lane];
    float* da = sall  + (size_t)b * H + lane * 4;
    float* dm = smask + (size_t)b * H + lane * 4;
    unsafeAtomicAdd(da + 0, v.x); unsafeAtomicAdd(da + 1, v.y);
    unsafeAtomicAdd(da + 2, v.z); unsafeAtomicAdd(da + 3, v.w);
    if (mw > 0.0f) {
        unsafeAtomicAdd(dm + 0, v.x); unsafeAtomicAdd(dm + 1, v.y);
        unsafeAtomicAdd(dm + 2, v.z); unsafeAtomicAdd(dm + 3, v.w);
    }
    if (lane == 0) {
        unsafeAtomicAdd(&cntc[b], 1.0f);
        if (mw > 0.0f) unsafeAtomicAdd(&cntm[b], 1.0f);
    }
}

// ---------------- kernel 7: gate + classifier, one block per graph ----------------
__device__ __forceinline__ float block_reduce_128(float v, float* red) {
    const int t = threadIdx.x;
    red[t] = v; __syncthreads();
    for (int s = 64; s > 0; s >>= 1) {
        if (t < s) red[t] += red[t + s];
        __syncthreads();
    }
    float r = red[0]; __syncthreads();
    return r;
}

__global__ __launch_bounds__(128)
void k_finalize(const float* __restrict__ fsum, const float* __restrict__ cntf,
                const float* __restrict__ smask, const float* __restrict__ sall,
                const float* __restrict__ cntm, const float* __restrict__ cntc,
                const float* __restrict__ Wgate, const float* __restrict__ bgate,
                const float* __restrict__ Wcls, const float* __restrict__ bcls,
                float* __restrict__ out) {
    __shared__ float red[128];
    const int b = blockIdx.x, t = threadIdx.x;
    const float cf = fmaxf(cntf[b], 1.0f);
    const float fp = fsum[(size_t)b * H + t] / cf;
    const float cm = cntm[b];
    const float cc = fmaxf(cntc[b], 1.0f);
    const float cp = (cm > 1e-6f)
        ? (smask[(size_t)b * H + t] / fmaxf(cm, 1.0f))
        : (sall[(size_t)b * H + t] / cc);
    const float gdot = block_reduce_128(Wgate[t] * fp + Wgate[H + t] * cp, red);
    const float alpha = sigmoidf(gdot + bgate[0]);
    const float pooled = alpha * fp + (1.0f - alpha) * cp;
    const float ldot = block_reduce_128(Wcls[t] * pooled, red);
    if (t == 0) out[b] = ldot + bcls[0];
}

// ---------------- host orchestration ----------------
extern "C" void kernel_launch(void* const* d_in, const int* in_sizes, int n_in,
                              void* d_out, int out_size, void* d_ws, size_t ws_size,
                              hipStream_t stream) {
    const float* x_fact  = (const float*)d_in[0];
    const float* x_comp  = (const float*)d_in[1];
    const float* ea_fc   = (const float*)d_in[2];
    const float* ea_cf   = (const float*)d_in[3];
    const int*   src_fc  = (const int*)d_in[4];
    const int*   dst_fc  = (const int*)d_in[5];
    const int*   src_cf  = (const int*)d_in[6];
    const int*   dst_cf  = (const int*)d_in[7];
    const int*   fbatch  = (const int*)d_in[8];
    const int*   cbatch  = (const int*)d_in[9];
    const unsigned char* pmask = (const unsigned char*)d_in[10];
    const float* W_in_f  = (const float*)d_in[11];
    const float* b_in_f  = (const float*)d_in[12];
    const float* g_in_f  = (const float*)d_in[13];
    const float* be_in_f = (const float*)d_in[14];
    const float* W_in_c  = (const float*)d_in[15];
    const float* b_in_c  = (const float*)d_in[16];
    const float* g_in_c  = (const float*)d_in[17];
    const float* be_in_c = (const float*)d_in[18];
    const float* w_mix   = (const float*)d_in[19];
    const float* b_mix   = (const float*)d_in[20];
    const float* Wrel_fc = (const float*)d_in[21];
    const float* brel_fc = (const float*)d_in[22];
    const float* Wroot_fc= (const float*)d_in[23];
    const float* Wrel_cf = (const float*)d_in[24];
    const float* brel_cf = (const float*)d_in[25];
    const float* Wroot_cf= (const float*)d_in[26];
    const float* g_post_f= (const float*)d_in[27];
    const float* b_post_f= (const float*)d_in[28];
    const float* g_post_c= (const float*)d_in[29];
    const float* b_post_c= (const float*)d_in[30];
    const float* W_gate  = (const float*)d_in[31];
    const float* b_gate  = (const float*)d_in[32];
    const float* W_cls   = (const float*)d_in[33];
    const float* b_cls   = (const float*)d_in[34];
    float* out = (float*)d_out;
    const int e = in_sizes[4];   // E

    // workspace layout (float units)
    float* ws = (float*)d_ws;
    const size_t NFH = (size_t)NF * H, NCH = (size_t)NC * H;
    float* hfA  = ws;            float* hfB  = hfA + NFH;
    float* msgF = hfB + NFH;
    float* hcA  = msgF + NFH;    float* hcB  = hcA + NCH;
    float* msgC = hcB + NCH;
    float* wfc  = msgC + NCH;    float* wcf  = wfc + e;
    float* fsum = wcf + e;       float* cntf = fsum + (size_t)B * H;
    float* smk  = cntf + B;      float* sall = smk + (size_t)B * H;
    float* cntm = sall + (size_t)B * H;  float* cntc = cntm + B;
    // bf16 weight areas (half-sized, expressed in float slots)
    unsigned short* WfBf      = (unsigned short*)(cntc + B);
    unsigned short* WrelFcBf  = WfBf     + (size_t)DF * H;
    unsigned short* WrootFcBf = WrelFcBf + (size_t)L * H * H;
    unsigned short* WrelCfBf  = WrootFcBf+ (size_t)L * H * H;
    unsigned short* WrootCfBf = WrelCfBf + (size_t)L * H * H;

    // one-time (per call) weight conversion to bf16
    const int nWf = DF * H, nWl = L * H * H;
    k_f32_to_bf16<<<(nWf / 4 + 255) / 256, 256, 0, stream>>>(W_in_f,  WfBf,      nWf);
    k_f32_to_bf16<<<(nWl / 4 + 255) / 256, 256, 0, stream>>>(Wrel_fc, WrelFcBf,  nWl);
    k_f32_to_bf16<<<(nWl / 4 + 255) / 256, 256, 0, stream>>>(Wroot_fc,WrootFcBf, nWl);
    k_f32_to_bf16<<<(nWl / 4 + 255) / 256, 256, 0, stream>>>(Wrel_cf, WrelCfBf,  nWl);
    k_f32_to_bf16<<<(nWl / 4 + 255) / 256, 256, 0, stream>>>(Wroot_cf,WrootCfBf, nWl);

    // encoders + gates
    k_encode_fact<<<NF / 16, 256, 0, stream>>>(x_fact, WfBf, b_in_f, g_in_f, be_in_f, hfA);
    k_encode_comp<<<NC, 128, 0, stream>>>(x_comp, W_in_c, b_in_c, g_in_c, be_in_c, hcA);
    k_edge_gate<<<(2 * e + 255) / 256, 256, 0, stream>>>(ea_fc, ea_cf, w_mix, b_mix, wfc, wcf, e);

    float* hf_cur = hfA; float* hf_nxt = hfB;
    float* hc_cur = hcA; float* hc_nxt = hcB;
    const int scat_blocks = (e * 32 + 255) / 256;
    for (int l = 0; l < L; ++l) {
        hipMemsetAsync(msgC, 0, NCH * sizeof(float), stream);
        hipMemsetAsync(msgF, 0, NFH * sizeof(float), stream);
        k_scatter<<<scat_blocks, 256, 0, stream>>>(src_fc, dst_fc, wfc, hf_cur, msgC, e);
        k_scatter<<<scat_blocks, 256, 0, stream>>>(src_cf, dst_cf, wcf, hc_cur, msgF, e);
        k_gnn_update<<<NC / 16, 256, 0, stream>>>(msgC, WrelFcBf + (size_t)l * H * H,
            brel_fc + (size_t)l * H, hc_cur, WrootFcBf + (size_t)l * H * H,
            g_post_c, b_post_c, hc_nxt);
        k_gnn_update<<<NF / 16, 256, 0, stream>>>(msgF, WrelCfBf + (size_t)l * H * H,
            brel_cf + (size_t)l * H, hf_cur, WrootCfBf + (size_t)l * H * H,
            g_post_f, b_post_f, hf_nxt);
        float* t;
        t = hf_cur; hf_cur = hf_nxt; hf_nxt = t;
        t = hc_cur; hc_cur = hc_nxt; hc_nxt = t;
    }

    // pooling accumulators: contiguous region fsum..cntc
    hipMemsetAsync(fsum, 0, (3 * (size_t)B * H + 3 * B) * sizeof(float), stream);
    k_pool_fact<<<(NF * 32 + 255) / 256, 256, 0, stream>>>(hf_cur, fbatch, fsum, cntf);
    k_pool_comp<<<(NC * 32 + 255) / 256, 256, 0, stream>>>(hc_cur, cbatch, pmask, smk, sall, cntm, cntc);
    k_finalize<<<B, 128, 0, stream>>>(fsum, cntf, smk, sall, cntm, cntc,
                                      W_gate, b_gate, W_cls, b_cls, out);
    (void)n_in; (void)out_size; (void)ws_size; (void)in_sizes;
}
